// GCN_16415365005351
// MI455X (gfx1250) — matrix operand order repaired and checked by reading, not compiled, not run on hardware
//
#include <hip/hip_runtime.h>
#include <hip/hip_bf16.h>
#include <math.h>

// ---------------------------------------------------------------------------
// GCN forward on gfx1250: WMMA f16 GEMMs (B staged in LDS via async copy)
// + atomic scatter aggregation (L2-resident) + wave-per-node LayerNorm.
// ---------------------------------------------------------------------------

typedef __attribute__((ext_vector_type(16))) _Float16 v16h;
typedef __attribute__((ext_vector_type(8)))  float    v8f;
typedef __attribute__((ext_vector_type(4)))  unsigned int v4u;

union FragAB { v16h h; v4u u2[2]; };

#define LN_EPS 1e-5f

// ----------------------- degree / normalization ----------------------------

__global__ void __launch_bounds__(256) deg_init(float* deg, int n) {
    int i = blockIdx.x * 256 + threadIdx.x;
    if (i < n) deg[i] = 1.0f;                 // self-loop contributes 1
}

__global__ void __launch_bounds__(256) deg_count(const int* __restrict__ dst,
                                                 float* __restrict__ deg, int E) {
    int i = blockIdx.x * 256 + threadIdx.x;
    if (i < E) atomicAdd(&deg[dst[i]], 1.0f);
}

__global__ void __launch_bounds__(256) deg_rsqrt(float* deg, int n) {
    int i = blockIdx.x * 256 + threadIdx.x;
    if (i < n) deg[i] = rsqrtf(deg[i]);       // deg >= 1 always (self-loop)
}

// ----------------------- conversions ---------------------------------------

__global__ void __launch_bounds__(256) f32_to_f16(const float* __restrict__ in,
                                                  _Float16* __restrict__ out, size_t n) {
    size_t i = (size_t)blockIdx.x * 256 + threadIdx.x;
    if (i < n) out[i] = (_Float16)in[i];
}

// W [K x NNin] f32 row-major  ->  Wt [NNout x K] f16 row-major (transposed, zero-pad)
__global__ void __launch_bounds__(256) transpose_pad_f16(const float* __restrict__ W,
                                                         _Float16* __restrict__ Wt,
                                                         int K, int NNin, int NNout) {
    int i = blockIdx.x * 256 + threadIdx.x;
    if (i >= NNout * K) return;
    int n = i / K, k = i - n * K;
    Wt[i] = (n < NNin) ? (_Float16)W[(size_t)k * NNin + n] : (_Float16)0.0f;
}

// ----------------------- WMMA GEMM -----------------------------------------
// C[M x NN] (f32) = A[M x 128] (f16 row-major) * Bt^T, Bt is [NN x 128] f16.
// Bt is staged into LDS once per block with async global->LDS copies
// (ASYNCcnt path), then each of the 8 waves computes a 16-row strip.
// All wave-uniform guards are broadcast to SGPRs (readfirstlane) so every
// WMMA executes with EXEC all-ones, as the ISA requires.

template <int NN>
__global__ void __launch_bounds__(256)
gemm_wmma_f16(const _Float16* __restrict__ A,
              const _Float16* __restrict__ Bt,
              float* __restrict__ C,
              int M)
{
    constexpr int K = 128;
    __shared__ _Float16 sB[NN * K];           // 32KB (NN=128) / 12KB (NN=48)

    const int tid  = threadIdx.x;
    const int lane = tid & 31;
    const int wid  = tid >> 5;

    // ---- stage Bt into LDS: async global->LDS, 16B per lane per iter ----
    // NN*16 chunks of 16B; NN is a multiple of 16 so every iter is full-EXEC.
    #pragma unroll
    for (int c = tid; c < NN * 16; c += 256) {
        unsigned lds_off = (unsigned)(size_t)(&sB[c * 8]);
        const _Float16* gp = Bt + (size_t)c * 8;
        asm volatile("global_load_async_to_lds_b128 %0, %1, off"
                     :: "v"(lds_off), "v"(gp) : "memory");
    }
    asm volatile("s_wait_asynccnt 0x0" ::: "memory");
    __syncthreads();

    const int mtiles = (M + 15) >> 4;
    // Broadcast wave-uniform tile id to an SGPR -> scalar branch, EXEC intact.
    const int stile = __builtin_amdgcn_readfirstlane(blockIdx.x * 8 + wid);
    if (stile < mtiles) {
        const int row0 = stile << 4;
        int arow = row0 + (lane & 15);
        if (arow >= M) arow = M - 1;          // clamp (ragged tail only)
        const int ksel = lane >> 4;           // 0 or 1 (half-wave K split)

        // A fragment: lane<16 holds K = ks*32 + {0..7,16..23}; lane>=16 +8.
        FragAB fa[4];
        #pragma unroll
        for (int ks = 0; ks < 4; ++ks) {
            const _Float16* ap = A + (size_t)arow * K + ks * 32 + ksel * 8;
            fa[ks].u2[0] = *(const v4u*)(ap);
            fa[ks].u2[1] = *(const v4u*)(ap + 16);
        }

        const bool fullTile = (row0 + 16 <= M);   // scalar (row0, M uniform)
        #pragma unroll
        for (int nt = 0; nt < NN / 16; ++nt) {
            const int col = (nt << 4) + (lane & 15);
            // Preload all 4 B fragments from LDS, then 4 back-to-back WMMAs.
            const _Float16* bp = &sB[col * K + ksel * 8];
            FragAB fb[4];
            #pragma unroll
            for (int ks = 0; ks < 4; ++ks) {
                fb[ks].u2[0] = *(const v4u*)(bp + ks * 32);
                fb[ks].u2[1] = *(const v4u*)(bp + ks * 32 + 16);
            }
            v8f acc = {};
            #pragma unroll
            for (int ks = 0; ks < 4; ++ks)
                acc = __builtin_amdgcn_wmma_f32_16x16x32_f16(
                          false, fa[ks].h, false, fb[ks].h,
                          (short)0, acc, false, false);

            // C/D layout: VGPR i holds M = ksel*8 + i, N = lane&15.
            float* cp = C + (size_t)(row0 + ksel * 8) * NN + (nt << 4) + (lane & 15);
            if (fullTile) {
                #pragma unroll
                for (int i = 0; i < 8; ++i) cp[(size_t)i * NN] = acc[i];
            } else {
                #pragma unroll
                for (int i = 0; i < 8; ++i)
                    if (row0 + ksel * 8 + i < M) cp[(size_t)i * NN] = acc[i];
            }
        }
    }
}

// ----------------------- edge aggregation ----------------------------------
// One wave per edge; each lane owns 4 consecutive features (128 total).
// agg (51MB) is L2-resident on MI455X (192MB L2): atomics resolve in L2.

__global__ void __launch_bounds__(256)
edge_agg128(const float* __restrict__ h, const int* __restrict__ src,
            const int* __restrict__ dst, const float* __restrict__ dinv,
            float* __restrict__ agg, int E)
{
    int e    = (blockIdx.x * 256 + threadIdx.x) >> 5;
    int lane = threadIdx.x & 31;
    if (e >= E) return;
    const int s = src[e], d = dst[e];
    const float nrm = dinv[s] * dinv[d];
    const float4 v = *(const float4*)(h + (size_t)s * 128 + lane * 4);
    float* ap = agg + (size_t)d * 128 + lane * 4;
    atomicAdd(ap + 0, v.x * nrm);
    atomicAdd(ap + 1, v.y * nrm);
    atomicAdd(ap + 2, v.z * nrm);
    atomicAdd(ap + 3, v.w * nrm);
}

// 48-wide (padded) version for the class layer.
__global__ void __launch_bounds__(256)
edge_agg48(const float* __restrict__ h, const int* __restrict__ src,
           const int* __restrict__ dst, const float* __restrict__ dinv,
           float* __restrict__ agg, int E)
{
    int e    = (blockIdx.x * 256 + threadIdx.x) >> 5;
    int lane = threadIdx.x & 31;
    if (e >= E) return;
    const int s = src[e], d = dst[e];
    const float nrm = dinv[s] * dinv[d];
    for (int f = lane; f < 48; f += 32)
        atomicAdd(&agg[(size_t)d * 48 + f], h[(size_t)s * 48 + f] * nrm);
}

// ----------------------- LayerNorm + ReLU  ->  f16 -------------------------
// One wave per node; lane owns 4 features. Adds self-loop term + bias first.

__global__ void __launch_bounds__(256)
post_ln_relu(const float* __restrict__ agg, const float* __restrict__ gout,
             const float* __restrict__ dinv, const float* __restrict__ b,
             const float* __restrict__ g, const float* __restrict__ be,
             _Float16* __restrict__ hf, int n)
{
    int node = (blockIdx.x * 256 + threadIdx.x) >> 5;
    int lane = threadIdx.x & 31;
    if (node >= n) return;
    const float di = dinv[node];
    const float sw = di * di;                 // self-loop norm dinv[i]^2
    const size_t base = (size_t)node * 128 + lane * 4;

    float v[4];
    #pragma unroll
    for (int j = 0; j < 4; ++j)
        v[j] = agg[base + j] + sw * gout[base + j] + b[lane * 4 + j];

    float s = v[0] + v[1] + v[2] + v[3];
    #pragma unroll
    for (int o = 16; o; o >>= 1) s += __shfl_xor(s, o, 32);
    const float mu = s * (1.0f / 128.0f);

    float q = 0.0f;
    #pragma unroll
    for (int j = 0; j < 4; ++j) { float dj = v[j] - mu; q += dj * dj; }
    #pragma unroll
    for (int o = 16; o; o >>= 1) q += __shfl_xor(q, o, 32);
    const float rstd = rsqrtf(q * (1.0f / 128.0f) + LN_EPS);

    #pragma unroll
    for (int j = 0; j < 4; ++j) {
        float y = (v[j] - mu) * rstd * g[lane * 4 + j] + be[lane * 4 + j];
        y = fmaxf(y, 0.0f);
        hf[base + j] = (_Float16)y;
    }
}

// ----------------------- final layer: bias + log_softmax -------------------
// One wave per node, 40 classes (stride-48 buffers).

__global__ void __launch_bounds__(256)
final_logsoftmax(const float* __restrict__ agg, const float* __restrict__ gout,
                 const float* __restrict__ dinv, const float* __restrict__ b3,
                 float* __restrict__ out, int n, int ncls)
{
    int node = (blockIdx.x * 256 + threadIdx.x) >> 5;
    int lane = threadIdx.x & 31;
    if (node >= n) return;
    const float di = dinv[node];
    const float sw = di * di;
    const size_t base = (size_t)node * 48;

    float v0 = -1e30f, v1 = -1e30f;
    const int c1 = lane + 32;
    if (lane < ncls) v0 = agg[base + lane] + sw * gout[base + lane] + b3[lane];
    if (c1   < ncls) v1 = agg[base + c1]   + sw * gout[base + c1]   + b3[c1];

    float m = fmaxf(v0, v1);
    #pragma unroll
    for (int o = 16; o; o >>= 1) m = fmaxf(m, __shfl_xor(m, o, 32));

    float e0 = (lane < ncls) ? expf(v0 - m) : 0.0f;
    float e1 = (c1   < ncls) ? expf(v1 - m) : 0.0f;
    float ss = e0 + e1;
    #pragma unroll
    for (int o = 16; o; o >>= 1) ss += __shfl_xor(ss, o, 32);
    const float lse = logf(ss);

    if (lane < ncls) out[(size_t)node * ncls + lane] = v0 - m - lse;
    if (c1   < ncls) out[(size_t)node * ncls + c1]   = v1 - m - lse;
}

// ---------------------------------------------------------------------------

extern "C" void kernel_launch(void* const* d_in, const int* in_sizes, int n_in,
                              void* d_out, int out_size, void* d_ws, size_t ws_size,
                              hipStream_t stream) {
    const float* x   = (const float*)d_in[0];
    const int*   ei  = (const int*)  d_in[1];
    const float* W1  = (const float*)d_in[2];
    const float* b1  = (const float*)d_in[3];
    const float* g1  = (const float*)d_in[4];
    const float* be1 = (const float*)d_in[5];
    const float* W2  = (const float*)d_in[6];
    const float* b2  = (const float*)d_in[7];
    const float* g2  = (const float*)d_in[8];
    const float* be2 = (const float*)d_in[9];
    const float* W3  = (const float*)d_in[10];
    const float* b3  = (const float*)d_in[11];
    float* out = (float*)d_out;

    const int F    = 128;
    const int N    = in_sizes[0] / F;         // 100000
    const int E    = in_sizes[1] / 2;         // 1600000
    const int NCLS = in_sizes[11];            // 40
    const int CP   = 48;                      // padded class width (3 WMMA n-tiles)

    // workspace carve-up
    char* ws = (char*)d_ws;
    size_t off = 0;
    auto carve = [&](size_t bytes) -> void* {
        void* p = ws + off;
        off += (bytes + 255) & ~(size_t)255;
        return p;
    };
    float*    dinv = (float*)   carve((size_t)N * 4);
    _Float16* hf   = (_Float16*)carve((size_t)N * F * 2);
    _Float16* wt   = (_Float16*)carve((size_t)F * F * 2);
    float*    gout = (float*)   carve((size_t)N * F * 4);
    float*    agg  = (float*)   carve((size_t)N * F * 4);
    (void)ws_size; (void)n_in; (void)out_size;

    const int* src = ei;
    const int* dst = ei + E;

    const int mtiles = (N + 15) / 16;
    dim3 blk(256);
    const int gN    = (N + 255) / 256;
    const int gE    = (E + 255) / 256;
    const int gEw   = (E + 7) / 8;            // wave-per-edge
    const int gNw   = (N + 7) / 8;            // wave-per-node
    const int gGemm = (mtiles + 7) / 8;

    // ---- gcn_norm: degrees with self loops, dinv = rsqrt(deg) ----
    deg_init <<<gN, blk, 0, stream>>>(dinv, N);
    deg_count<<<gE, blk, 0, stream>>>(dst, dinv, E);
    deg_rsqrt<<<gN, blk, 0, stream>>>(dinv, N);

    // ---- layer 1 ----
    f32_to_f16<<<(int)(((size_t)N * F + 255) / 256), blk, 0, stream>>>(x, hf, (size_t)N * F);
    transpose_pad_f16<<<(F * F + 255) / 256, blk, 0, stream>>>(W1, wt, F, F, F);
    gemm_wmma_f16<128><<<gGemm, blk, 0, stream>>>(hf, wt, gout, N);
    hipMemsetAsync(agg, 0, (size_t)N * F * 4, stream);
    edge_agg128<<<gEw, blk, 0, stream>>>(gout, src, dst, dinv, agg, E);
    post_ln_relu<<<gNw, blk, 0, stream>>>(agg, gout, dinv, b1, g1, be1, hf, N);

    // ---- layer 2 ----
    transpose_pad_f16<<<(F * F + 255) / 256, blk, 0, stream>>>(W2, wt, F, F, F);
    gemm_wmma_f16<128><<<gGemm, blk, 0, stream>>>(hf, wt, gout, N);
    hipMemsetAsync(agg, 0, (size_t)N * F * 4, stream);
    edge_agg128<<<gEw, blk, 0, stream>>>(gout, src, dst, dinv, agg, E);
    post_ln_relu<<<gNw, blk, 0, stream>>>(agg, gout, dinv, b2, g2, be2, hf, N);

    // ---- layer 3: 128 -> 40 (padded to 48) + log_softmax ----
    transpose_pad_f16<<<(CP * F + 255) / 256, blk, 0, stream>>>(W3, wt, F, NCLS, CP);
    gemm_wmma_f16<48><<<gGemm, blk, 0, stream>>>(hf, wt, gout, N);
    hipMemsetAsync(agg, 0, (size_t)N * CP * 4, stream);
    edge_agg48<<<gEw, blk, 0, stream>>>(gout, src, dst, dinv, agg, E);
    final_logsoftmax<<<gNw, blk, 0, stream>>>(agg, gout, dinv, b3, out, N, NCLS);
}
